// KBAModule_721554505978
// MI455X (gfx1250) — matrix operand order repaired and checked by reading, not compiled, not run on hardware
//
#include <hip/hip_runtime.h>

// Problem sizes (fixed by the reference)
#define Bz  8
#define Cch 64
#define Hh  256
#define Ww  256
#define Kk  32
#define BDIM 256   // 8 waves of 32

typedef __bf16 bf16_t;
typedef bf16_t v16bf __attribute__((ext_vector_type(16)));
typedef float  v8f   __attribute__((ext_vector_type(8)));

union FragU { v16bf v; uint4 q[2]; };

// ---- LDS layout (bytes, all 16B aligned) ----
#define OFF_W1    0          // [9][32][64] bf16  = 36864
#define OFF_W2    36864      // [9][32][32] bf16  = 18432
#define OFF_KB    55296      // [9][32][64] bf16  = 36864
#define OFF_WT    92160      // [64][64]   bf16   = 8192
#define OFF_BIAS  100352     // 128 f32           = 512
#define OFF_XS    100864     // [20*20][64] bf16  = 51200
#define OFF_XTS   152064     // [18*18][64] bf16  = 41472
#define OFF_OUT1  193536     // 256 f32           = 1024
#define OFF_STAGE 194560     // 25600 f32 staging = 102400 (reused below)
#define OFF_FC1   OFF_STAGE             // [18*18][32] bf16 = 20736
#define OFF_FC2   (OFF_STAGE + 20736)   // [16*16][32] bf16 = 16384
#define SMEM_BYTES 296960

__device__ __forceinline__ unsigned short f2bf(float f) {
  unsigned u = __builtin_bit_cast(unsigned, f);
  u += 0x7FFFu + ((u >> 16) & 1u);          // round-to-nearest-even
  return (unsigned short)(u >> 16);
}
__device__ __forceinline__ float bf2f(unsigned short h) {
  unsigned u = ((unsigned)h) << 16;
  return __builtin_bit_cast(float, u);
}

// One 16x16x32 bf16 WMMA step. A-fragment = two 16B runs (K pattern
// {0..7,16..23}+8*khalf per ISA layout), B-fragment = two 16B runs
// (same K pattern, N = lane&15).
__device__ __forceinline__ v8f wmma_step(const unsigned short* a0,
                                         const unsigned short* a1,
                                         const unsigned short* b0, v8f acc) {
  FragU fa, fb;
  fa.q[0] = *(const uint4*)a0;
  fa.q[1] = *(const uint4*)a1;
  fb.q[0] = ((const uint4*)b0)[0];
  fb.q[1] = ((const uint4*)b0)[2];
  return __builtin_amdgcn_wmma_f32_16x16x32_bf16(
      false, fa.v, false, fb.v, (short)0, acc, false, false);
}

__global__ __launch_bounds__(BDIM, 1)
void kba_fused_kernel(const float* __restrict__ x,
                      const float* __restrict__ kb,
                      const float* __restrict__ w1, const float* __restrict__ b1,
                      const float* __restrict__ w2, const float* __restrict__ b2,
                      const float* __restrict__ wt, const float* __restrict__ bt,
                      float* __restrict__ out) {
  extern __shared__ __align__(16) unsigned char smem[];
  unsigned short* w1_s  = (unsigned short*)(smem + OFF_W1);   // [tap][k][c]
  unsigned short* w2_s  = (unsigned short*)(smem + OFF_W2);   // [tap][k][c]
  unsigned short* kb_s  = (unsigned short*)(smem + OFF_KB);   // [tap][k][c]
  unsigned short* wt_s  = (unsigned short*)(smem + OFF_WT);   // [k][c]
  float*          bias_s= (float*)(smem + OFF_BIAS);          // b1|b2|bt
  unsigned short* x_s   = (unsigned short*)(smem + OFF_XS);   // [pix20][c]
  unsigned short* xt_s  = (unsigned short*)(smem + OFF_XTS);  // [pix18][c]
  float*          out1_s= (float*)(smem + OFF_OUT1);
  float*          stage = (float*)(smem + OFF_STAGE);         // fp32 x staging
  unsigned short* fc1_s = (unsigned short*)(smem + OFF_FC1);  // [pix18][k]
  unsigned short* fc2_s = (unsigned short*)(smem + OFF_FC2);  // [pix16][k]

  const int tid   = threadIdx.x;
  const int lane  = tid & 31;
  const int wv    = tid >> 5;
  const int m16   = lane & 15;
  const int khalf = lane >> 4;

  const int w0 = blockIdx.x * 16;
  const int h0 = blockIdx.y * 16;
  const int b  = blockIdx.z;

  // ---- load weights fp32->bf16 into [tap][k][c] layouts ----
  for (int i = tid; i < 18432; i += BDIM) {            // w1 (32,64,3,3)
    int kk = i / 576, rem = i % 576, c = rem / 9, tap = rem % 9;
    w1_s[(tap * 32 + kk) * 64 + c] = f2bf(w1[i]);
  }
  for (int i = tid; i < 9216; i += BDIM) {             // w2 (32,32,3,3)
    int kk = i / 288, rem = i % 288, c = rem / 9, tap = rem % 9;
    w2_s[(tap * 32 + kk) * 32 + c] = f2bf(w2[i]);
  }
  for (int i = tid; i < 18432; i += BDIM) {            // kernel_bases (32,64,3,3)
    int kk = i / 576, rem = i % 576, c = rem / 9, tap = rem % 9;
    kb_s[(tap * 32 + kk) * 64 + c] = f2bf(kb[i]);
  }
  for (int i = tid; i < 4096; i += BDIM) wt_s[i] = f2bf(wt[i]);  // (64,64,1,1)
  if (tid < 32)        bias_s[tid] = b1[tid];
  else if (tid < 64)   bias_s[tid] = b2[tid - 32];
  else if (tid < 128)  bias_s[tid] = bt[tid - 64];

  // ---- zero fp32 staging (covers zero-padding for halo) ----
  for (int i = tid; i < 25600; i += BDIM) stage[i] = 0.f;
  out1_s[tid] = 0.f;
  __syncthreads();

  // ---- async copy x tile (20x20 halo, 64ch) global -> LDS (ASYNCcnt) ----
  const unsigned long long xb64 =
      (unsigned long long)(size_t)(x + (size_t)b * (Cch * Hh * Ww));
  for (int i = tid; i < 25600; i += BDIM) {
    int c = i / 400, r = i % 400, iy = r / 20, ix = r % 20;
    int gy = h0 + iy - 2, gx = w0 + ix - 2;
    if (gy >= 0 && gy < Hh && gx >= 0 && gx < Ww) {
      unsigned voff = (unsigned)((((c * Hh) + gy) * Ww + gx) * 4);
      unsigned ldsa = (unsigned)(size_t)(stage + i);
      asm volatile("global_load_async_to_lds_b32 %0, %1, %2 offset:0"
                   :: "v"(ldsa), "v"(voff), "s"(xb64) : "memory");
    }
  }
  asm volatile("s_wait_asynccnt 0" ::: "memory");
  __syncthreads();

  // ---- convert staging fp32 -> bf16 [pixel][channel] ----
  for (int i = tid; i < 25600; i += BDIM) {
    int c = i / 400, r = i % 400;
    x_s[r * 64 + c] = f2bf(stage[i]);
  }
  __syncthreads();

  // ---- stage 1: conv1 (3x3, C=64 -> K=32) + bias + ReLU over 18x18 ----
  // implicit GEMM: M = 324 pixels (21 Mtiles padded), N = 32 (2 Ntiles),
  // K-loop = 2 channel chunks x 9 taps
  for (int t = wv; t < 42; t += 8) {
    int Mt = t >> 1, Nt = t & 1;
    int p = Mt * 16 + m16; if (p > 323) p = 323;
    int py = p / 18, px = p % 18;
    v8f acc = {0.f, 0.f, 0.f, 0.f, 0.f, 0.f, 0.f, 0.f};
    for (int cc = 0; cc < 2; ++cc) {
#pragma unroll
      for (int tap = 0; tap < 9; ++tap) {
        int dy = tap / 3, dx = tap % 3;
        const unsigned short* arow = x_s + ((py + dy) * 20 + (px + dx)) * 64;
        const unsigned short* a0 = arow + cc * 32 + khalf * 8;
        const unsigned short* b0 =
            w1_s + (tap * 32 + Nt * 16 + m16) * 64 + cc * 32 + khalf * 8;
        acc = wmma_step(a0, a0 + 16, b0, acc);
      }
    }
    int k = Nt * 16 + m16;
    float bias = bias_s[k];
#pragma unroll
    for (int r = 0; r < 8; ++r) {
      int pm = Mt * 16 + r + 8 * khalf;
      if (pm < 324) {
        float v = acc[r] + bias;
        fc1_s[pm * 32 + k] = f2bf(v > 0.f ? v : 0.f);
      }
    }
  }

  // ---- stage 2: xt = 1x1 transform (C=64 -> C=64) + bias over 18x18 ----
  for (int t = wv; t < 84; t += 8) {
    int Mt = t >> 2, Nt = t & 3;
    int p = Mt * 16 + m16; if (p > 323) p = 323;
    int py = p / 18, px = p % 18;
    const unsigned short* arow = x_s + ((py + 1) * 20 + (px + 1)) * 64;
    v8f acc = {0.f, 0.f, 0.f, 0.f, 0.f, 0.f, 0.f, 0.f};
#pragma unroll
    for (int cc = 0; cc < 2; ++cc) {
      const unsigned short* a0 = arow + cc * 32 + khalf * 8;
      const unsigned short* b0 =
          wt_s + (Nt * 16 + m16) * 64 + cc * 32 + khalf * 8;
      acc = wmma_step(a0, a0 + 16, b0, acc);
    }
    int k = Nt * 16 + m16;
    float bias = bias_s[64 + k];
#pragma unroll
    for (int r = 0; r < 8; ++r) {
      int pm = Mt * 16 + r + 8 * khalf;
      if (pm < 324) xt_s[pm * 64 + k] = f2bf(acc[r] + bias);
    }
  }
  __syncthreads();

  // ---- stage 3: conv2 (3x3, K=32 -> K=32) + bias over 16x16 ----
  for (int t = wv; t < 32; t += 8) {
    int Mt = t >> 1, Nt = t & 1;
    int p = Mt * 16 + m16;
    int py = p / 16, px = p % 16;
    v8f acc = {0.f, 0.f, 0.f, 0.f, 0.f, 0.f, 0.f, 0.f};
#pragma unroll
    for (int tap = 0; tap < 9; ++tap) {
      int dy = tap / 3, dx = tap % 3;
      const unsigned short* arow = fc1_s + ((py + dy) * 18 + (px + dx)) * 32;
      const unsigned short* a0 = arow + khalf * 8;
      const unsigned short* b0 =
          w2_s + (tap * 32 + Nt * 16 + m16) * 32 + khalf * 8;
      acc = wmma_step(a0, a0 + 16, b0, acc);
    }
    int k = Nt * 16 + m16;
    float bias = bias_s[32 + k];
#pragma unroll
    for (int r = 0; r < 8; ++r) {
      int pm = Mt * 16 + r + 8 * khalf;
      fc2_s[pm * 32 + k] = f2bf(acc[r] + bias);
    }
  }
  __syncthreads();

  // ---- stage 4: y = conv(xt, kernel_bases) fused with sum_k(y * fc2) ----
  for (int t = wv; t < 32; t += 8) {
    int Mt = t >> 1, Nt = t & 1;
    int p = Mt * 16 + m16;
    int py = p / 16, px = p % 16;
    v8f acc = {0.f, 0.f, 0.f, 0.f, 0.f, 0.f, 0.f, 0.f};
    for (int cc = 0; cc < 2; ++cc) {
#pragma unroll
      for (int tap = 0; tap < 9; ++tap) {
        int dy = tap / 3, dx = tap % 3;
        const unsigned short* arow = xt_s + ((py + dy) * 18 + (px + dx)) * 64;
        const unsigned short* a0 = arow + cc * 32 + khalf * 8;
        const unsigned short* b0 =
            kb_s + (tap * 32 + Nt * 16 + m16) * 64 + cc * 32 + khalf * 8;
        acc = wmma_step(a0, a0 + 16, b0, acc);
      }
    }
    int k = Nt * 16 + m16;
#pragma unroll
    for (int r = 0; r < 8; ++r) {
      int pm = Mt * 16 + r + 8 * khalf;
      atomicAdd(&out1_s[pm], acc[r] * bf2f(fc2_s[pm * 32 + k]));  // ds_add_f32
    }
  }
  __syncthreads();

  // ---- broadcast out1 across all 64 channels ----
  {
    float val = out1_s[tid];
    int py = tid / 16, px = tid % 16;
    size_t base = (size_t)b * Cch * Hh * Ww + (size_t)(h0 + py) * Ww + (w0 + px);
#pragma unroll 4
    for (int c = 0; c < Cch; ++c) out[base + (size_t)c * Hh * Ww] = val;
  }
}

extern "C" void kernel_launch(void* const* d_in, const int* in_sizes, int n_in,
                              void* d_out, int out_size, void* d_ws, size_t ws_size,
                              hipStream_t stream) {
  (void)in_sizes; (void)n_in; (void)d_ws; (void)ws_size; (void)out_size;
  const float* x  = (const float*)d_in[0];
  const float* kb = (const float*)d_in[1];
  const float* w1 = (const float*)d_in[2];
  const float* b1 = (const float*)d_in[3];
  const float* w2 = (const float*)d_in[4];
  const float* b2 = (const float*)d_in[5];
  const float* wt = (const float*)d_in[6];
  const float* bt = (const float*)d_in[7];
  float* out = (float*)d_out;

  dim3 grid(Ww / 16, Hh / 16, Bz);
  hipLaunchKernelGGL(kba_fused_kernel, grid, dim3(BDIM), SMEM_BYTES, stream,
                     x, kb, w1, b1, w2, b2, wt, bt, out);
}
